// _CNN_Geometric_Backflow_GPU_82978768159518
// MI455X (gfx1250) — matrix operand order repaired and checked by reading, not compiled, not run on hardware
//
#include <hip/hip_runtime.h>
#include <hip/hip_bf16.h>

typedef __bf16 bf16;
typedef __bf16 bf16x8  __attribute__((ext_vector_type(8)));
typedef __bf16 bf16x16 __attribute__((ext_vector_type(16)));
typedef float  f32x8   __attribute__((ext_vector_type(8)));
typedef unsigned int u32x4 __attribute__((ext_vector_type(4)));
typedef unsigned int u32x8 __attribute__((ext_vector_type(8)));

#define NS     256   // 16x16 sites
#define BATCH  256
#define HCH    256   // hidden channels
#define C0P    64    // 34 input channels padded to 64 for K%32==0
#define OUT_PER_B 107648

// LDS geometry (elements of bf16)
#define APAD   40    // row stride: 32 K-elems + 4-dword pad (80B: 16B-aligned, bank-clean)
#define ABUF   (128 * APAD)           // 5120 elems
#define BBUF   (64 * APAD)            // 2560 elems
#define CBUF   (ABUF + BBUF)          // 7680 elems per pipeline slot
#define NSLOT  3                      // TDM pipeline depth
#define HPAD   264   // head LDS row stride: 256 + 8 pad (528B: 16B-aligned, bank-clean)

extern __shared__ char cdna5_smem[];

// ---------------------------------------------------------------- helpers

__device__ __forceinline__ float gelu_exact(float v) {
    return 0.5f * v * (1.0f + erff(v * 0.7071067811865476f));
}

// A-fragment (16x32 bf16, MxK): elements 0..7 <- p[0..7], 8..15 <- p[16..23]
__device__ __forceinline__ bf16x16 frag_gap16(const bf16* __restrict__ p) {
    bf16x8 lo = *reinterpret_cast<const bf16x8*>(p);
    bf16x8 hi = *reinterpret_cast<const bf16x8*>(p + 16);
    return __builtin_shufflevector(lo, hi, 0,1,2,3,4,5,6,7,8,9,10,11,12,13,14,15);
}
// B-fragment (32x16 bf16, KxN): 16 contiguous elements
__device__ __forceinline__ bf16x16 frag_contig(const bf16* __restrict__ p) {
    bf16x8 lo = *reinterpret_cast<const bf16x8*>(p);
    bf16x8 hi = *reinterpret_cast<const bf16x8*>(p + 8);
    return __builtin_shufflevector(lo, hi, 0,1,2,3,4,5,6,7,8,9,10,11,12,13,14,15);
}

// CDNA5 async DMA: global -> LDS, 16B per lane, tracked by ASYNCcnt.
__device__ __forceinline__ void async_b128(unsigned lds_off, const bf16* g) {
    asm volatile("global_load_async_to_lds_b128 %0, %1, off"
                 :: "v"(lds_off), "v"((unsigned long long)(uintptr_t)g)
                 : "memory");
}
__device__ __forceinline__ void wait_async0() {
    asm volatile("s_wait_asynccnt 0x0" ::: "memory");
}

// ---- Tensor Data Mover: 2-D tile DMA global->LDS, tracked by TENSORcnt ----
// D# group1 for a [nrows x 32 bf16] tile, row stride `stride` elems,
// HW pad: 4 dwords after every 16 dwords -> LDS row pitch = APAD elems.
__device__ __forceinline__ u32x8 tdm_g1(unsigned nrows, unsigned stride) {
    u32x8 g;
    g[0] = (1u << 16)          // data_size = 1 (2-byte elements)
         | (1u << 20)          // pad_enable
         | (3u << 22)          // pad_interval: 16 dwords (64B row)
         | (3u << 25);         // pad_amount: 4 dwords (16B)
    g[1] = 32u << 16;          // tensor_dim0 = 32 (bits 79:48 low half)
    g[2] = nrows << 16;        // tensor_dim1 (bits 111:80 low half)
    g[3] = 32u << 16;          // tile_dim0 = 32 (bits 127:112)
    g[4] = nrows;              // tile_dim1 (bits 143:128); tile_dim2 = 0
    g[5] = stride;             // tensor_dim0_stride[31:0] (bits 207:160)
    g[6] = 0u;                 // stride hi, dim1_stride lo
    g[7] = 0u;
    return g;
}
__device__ __forceinline__ void tdm_load(unsigned lds_bytes, const bf16* g, u32x8 g1) {
    unsigned long long ga = (unsigned long long)(uintptr_t)g;
    u32x4 g0;
    g0[0] = 1u;                                            // count = 1 valid descriptor
    g0[1] = lds_bytes;                                     // lds_addr
    g0[2] = (unsigned)ga;                                  // global_addr[31:0]
    g0[3] = (unsigned)((ga >> 32) & 0x1FFFFFFu) | (2u << 30); // addr[56:32] | type=2
    asm volatile("tensor_load_to_lds %0, %1" :: "s"(g0), "s"(g1) : "memory");
}

// ---------------------------------------------------------------- prep kernels

__global__ __launch_bounds__(256) void k_embed(const int* __restrict__ x,
                                               const float* __restrict__ emb,
                                               bf16* __restrict__ in0) {
    int t = blockIdx.x * 256 + threadIdx.x;   // (b, s)
    int s = t & 255;
    int b = t >> 8;
    int xv = x[b * NS + s];
    bf16* dst = in0 + ((size_t)s * BATCH + b) * C0P;
    const float* e = emb + xv * 32;
#pragma unroll
    for (int c = 0; c < 32; ++c) dst[c] = (bf16)e[c];
    int xi = s >> 4, yi = s & 15;
    dst[32] = (bf16)(-1.0f + (2.0f / 15.0f) * (float)xi);
    dst[33] = (bf16)(-1.0f + (2.0f / 15.0f) * (float)yi);
#pragma unroll
    for (int c = 34; c < C0P; ++c) dst[c] = (bf16)0.0f;
}

__global__ __launch_bounds__(256) void k_prep_w0(const float* __restrict__ src,
                                                 bf16* __restrict__ dst) {
    int e = blockIdx.x * 256 + threadIdx.x;          // 9*256*64
    int tap = e >> 14;
    int r   = e & 16383;
    int co  = r >> 6;
    int ci  = r & 63;
    dst[e] = (ci < 34) ? (bf16)src[(co * 34 + ci) * 9 + tap] : (bf16)0.0f;
}

__global__ __launch_bounds__(256) void k_prep_w1(const float* __restrict__ src,
                                                 bf16* __restrict__ dst) {
    int e = blockIdx.x * 256 + threadIdx.x;          // 9*256*256
    int tap = e >> 16;
    int r   = e & 65535;
    int co  = r >> 8;
    int ci  = r & 255;
    dst[e] = (bf16)src[(co * 256 + ci) * 9 + tap];
}

__global__ __launch_bounds__(256) void k_cvt(const float* __restrict__ src,
                                             bf16* __restrict__ dst, int n) {
    int i = blockIdx.x * 256 + threadIdx.x;
    if (i < n) dst[i] = (bf16)src[i];
}

// ---------------------------------------------------------------- conv (implicit GEMM)
// act: [site][batch][ch] bf16; weights: [tap][co][ci] bf16.
// grid: (256 sites, 8): y&1 -> batch half (128), y>>1 -> co quarter (64).
// block: 8 waves; wave computes 16(b) x 64(co) = 4 accumulators.
// Per 32-K chunk: wave0 issues TWO TDM tile loads (A 128x32, B 64x32) into a
// 3-deep LDS pipeline; waits are partial (TENSORcnt <= 2) so 2 chunks of DMA
// stay in flight behind the WMMAs.

template <int CINP>
__global__ __launch_bounds__(256) void k_conv(const bf16* __restrict__ in,
                                              const bf16* __restrict__ w,
                                              const float* __restrict__ bias,
                                              bf16* __restrict__ out) {
    constexpr int KCPT = CINP / 32;            // K chunks per tap
    constexpr int KSH  = (CINP == 64) ? 1 : 3; // log2(KCPT)

    const int s    = blockIdx.x;
    const int b0   = (blockIdx.y & 1) * 128;
    const int co0  = (blockIdx.y >> 1) * 64;
    const int tid  = threadIdx.x;
    const int wave = tid >> 5;
    const int lane = tid & 31;
    const int m = lane & 15;
    const int h = lane >> 4;
    const int n = m;
    const int xi = s >> 4, yi = s & 15;

    // valid taps, bit-packed (12b: neighbor site | tap<<8) to stay in SGPRs
    unsigned long long pk0 = 0, pk1 = 0;
    int ntaps = 0;
    for (int i = 0; i < 3; ++i) {
        int nx = xi + i - 1;
        if (nx < 0 || nx > 15) continue;
        for (int j = 0; j < 3; ++j) {
            int ny = yi + j - 1;
            if (ny < 0 || ny > 15) continue;
            unsigned e = (unsigned)(((nx << 4) | ny) | ((i * 3 + j) << 8));
            if (ntaps < 5) pk0 |= (unsigned long long)e << (12 * ntaps);
            else           pk1 |= (unsigned long long)e << (12 * (ntaps - 5));
            ++ntaps;
        }
    }
    const int nchunks = ntaps * KCPT;          // >= 8 always

    bf16* sm = (bf16*)cdna5_smem;
    const u32x8 g1A = tdm_g1(128u, (unsigned)CINP);
    const u32x8 g1B = tdm_g1(64u,  (unsigned)CINP);

    auto issue = [&](int c, int slot) {        // wave0 only
        int ti = c >> KSH;
        int k0 = (c & (KCPT - 1)) << 5;
        unsigned e = (unsigned)((ti < 5 ? (pk0 >> (12 * ti))
                                        : (pk1 >> (12 * (ti - 5)))) & 0xFFF);
        int nsit = e & 255;
        int tap  = e >> 8;
        unsigned lbase = 2u * (unsigned)(slot * CBUF);
        tdm_load(lbase,
                 in + ((size_t)nsit * BATCH + b0) * CINP + k0, g1A);
        tdm_load(lbase + 2u * ABUF,
                 w + ((size_t)tap * HCH + co0) * CINP + k0, g1B);
    };

    f32x8 acc0 = {}, acc1 = {}, acc2 = {}, acc3 = {};

    if (wave == 0) { issue(0, 0); issue(1, 1); }
    __builtin_amdgcn_s_wait_tensorcnt(2);      // chunk 0 landed
    __syncthreads();

    int pc = 0, pi = 2;
    for (int c = 0; c < nchunks; ++c) {
        const bool more = (c + 2) < nchunks;
        if (more) {
            if (wave == 0) issue(c + 2, pi);
            pi = (pi == NSLOT - 1) ? 0 : pi + 1;
        }
        // compute chunk c from slot pc: distinct frags so ds_loads clause up
        const bf16* Arow = sm + pc * CBUF + (wave * 16 + m) * APAD + 8 * h;
        const bf16* Bb   = sm + pc * CBUF + ABUF + n * APAD + 8 * h;
        bf16x16 a  = frag_gap16(Arow);
        bf16x16 f0 = frag_gap16(Bb);
        bf16x16 f1 = frag_gap16(Bb + 16 * APAD);
        bf16x16 f2 = frag_gap16(Bb + 32 * APAD);
        bf16x16 f3 = frag_gap16(Bb + 48 * APAD);
        acc0 = __builtin_amdgcn_wmma_f32_16x16x32_bf16(false, a, false, f0, (short)0, acc0, false, false);
        acc1 = __builtin_amdgcn_wmma_f32_16x16x32_bf16(false, a, false, f1, (short)0, acc1, false, false);
        acc2 = __builtin_amdgcn_wmma_f32_16x16x32_bf16(false, a, false, f2, (short)0, acc2, false, false);
        acc3 = __builtin_amdgcn_wmma_f32_16x16x32_bf16(false, a, false, f3, (short)0, acc3, false, false);
        pc = (pc == NSLOT - 1) ? 0 : pc + 1;
        if (more) __builtin_amdgcn_s_wait_tensorcnt(2);  // chunk c+1 landed
        else      __builtin_amdgcn_s_wait_tensorcnt(0);
        __syncthreads();
    }

    const int brow0 = b0 + wave * 16;
    f32x8 accs[4] = {acc0, acc1, acc2, acc3};
#pragma unroll
    for (int cc = 0; cc < 4; ++cc) {
        float bz = bias[co0 + cc * 16 + n];
#pragma unroll
        for (int v = 0; v < 8; ++v) {
            float val = gelu_exact(accs[cc][v] + bz);
            int brow = brow0 + v + 8 * h;
            out[((size_t)s * BATCH + brow) * HCH + co0 + cc * 16 + n] = (bf16)val;
        }
    }
}

// ---------------------------------------------------------------- heads
__device__ __constant__ int g_wboff[9] = {0, 8192, 40960, 49152, 81920,
                                          212992, 245760, 253952, 286720};

struct HeadBias { const float* p[9]; };

__global__ __launch_bounds__(256) void k_head(const bf16* __restrict__ act,
                                              const bf16* __restrict__ hw,
                                              HeadBias hb,
                                              float* __restrict__ out) {
    const int s    = blockIdx.x;
    const int b0   = blockIdx.y * 16;
    const int tid  = threadIdx.x;
    const int wave = tid >> 5;
    const int lane = tid & 31;
    const int m = lane & 15;
    const int h = lane >> 4;
    const int n = m;
    const int xi = s >> 4, yi = s & 15;

    const bool tx = (xi == 0), bx = (xi == 15), ly = (yi == 0), ry = (yi == 15);
    int g;
    if (tx && ly) g = 0; else if (tx && ry) g = 2;
    else if (bx && ly) g = 6; else if (bx && ry) g = 8;
    else if (tx) g = 1; else if (bx) g = 7;
    else if (ly) g = 3; else if (ry) g = 5; else g = 4;

    const int nbound = (int)(tx || bx) + (int)(ly || ry);
    const int od = (nbound == 2) ? 32 : (nbound == 1 ? 128 : 512);

    const bf16* wbase = hw + g_wboff[g];
    const float* bias = hb.p[g];

    int rowbase = (xi == 0) ? 0 : 1856 + (xi - 1) * 7424;
    int partial;
    if (xi == 0 || xi == 15) partial = (yi == 0) ? 0 : 32 + (yi - 1) * 128;
    else                     partial = (yi == 0) ? 0 : 128 + (yi - 1) * 512;
    const size_t outoff = (size_t)rowbase + partial;

    // async-DMA the 16x256 activation tile into LDS once; shared by all N-tiles.
    bf16* sm = (bf16*)cdna5_smem;
    {
        int r = tid >> 4, q = tid & 15;     // 16 rows, 16 threads/row, 32B each
        const bf16* ga = act + ((size_t)s * BATCH + b0 + r) * HCH + q * 16;
        unsigned    la = 2u * (unsigned)(r * HPAD + q * 16);
        async_b128(la,      ga);
        async_b128(la + 16, ga + 8);
    }
    wait_async0();
    __syncthreads();

    const bf16* Arow = sm + m * HPAD;

    for (int nt = wave; nt * 16 < od; nt += 8) {
        int n0 = nt * 16;
        f32x8 acc = {};
        const bf16* bcol = wbase + (size_t)(n0 + n) * HCH + 8 * h;
        __builtin_prefetch(wbase + (size_t)(n0 + 128 + n) * HCH, 0, 1);
#pragma unroll
        for (int k0 = 0; k0 < HCH; k0 += 32) {
            bf16x16 a  = frag_gap16(Arow + k0 + 8 * h);
            bf16x16 bm = frag_gap16(bcol + k0);
            acc = __builtin_amdgcn_wmma_f32_16x16x32_bf16(
                false, a, false, bm, (short)0, acc, false, false);
        }
        float bz = bias[n0 + n];
#pragma unroll
        for (int v = 0; v < 8; ++v) {
            int b = b0 + v + 8 * h;
            out[(size_t)b * OUT_PER_B + outoff + n0 + n] = acc[v] + bz;
        }
    }
}

// ---------------------------------------------------------------- launch

extern "C" void kernel_launch(void* const* d_in, const int* in_sizes, int n_in,
                              void* d_out, int out_size, void* d_ws, size_t ws_size,
                              hipStream_t stream) {
    (void)in_sizes; (void)n_in; (void)out_size; (void)ws_size;

    const int*   x       = (const int*)  d_in[0];
    const float* emb     = (const float*)d_in[1];
    const float* conv_w0 = (const float*)d_in[2];
    const float* conv_b0 = (const float*)d_in[3];
    const float* conv_w1 = (const float*)d_in[4];
    const float* conv_b1 = (const float*)d_in[5];

    char* ws = (char*)d_ws;
    bf16* in0  = (bf16*)(ws + 0);                        //  8,388,608 B
    bf16* w0b  = (bf16*)(ws + 8388608);                  //    294,912 B
    bf16* w1b  = (bf16*)(ws + 8683520);                  //  1,179,648 B
    bf16* hwb  = (bf16*)(ws + 9863168);                  //    589,824 B
    bf16* act0 = (bf16*)(ws + 10452992);                 // 33,554,432 B
    bf16* act1 = (bf16*)(ws + 44007424);                 // 33,554,432 B

    float* out = (float*)d_out;

    k_embed  <<<256,  256, 0, stream>>>(x, emb, in0);
    k_prep_w0<<<576,  256, 0, stream>>>(conv_w0, w0b);
    k_prep_w1<<<2304, 256, 0, stream>>>(conv_w1, w1b);

    static const int od_tab[9] = {32, 128, 32, 128, 512, 128, 32, 128, 32};
    static const int wb_tab[9] = {0, 8192, 40960, 49152, 81920,
                                  212992, 245760, 253952, 286720};
    HeadBias hb;
    for (int g = 0; g < 9; ++g) {
        const float* hwsrc = (const float*)d_in[6 + 2 * g];
        hb.p[g]            = (const float*)d_in[7 + 2 * g];
        int nelem = od_tab[g] * HCH;
        k_cvt<<<(nelem + 255) / 256, 256, 0, stream>>>(hwsrc, hwb + wb_tab[g], nelem);
    }

    const size_t conv_lds = (size_t)NSLOT * CBUF * sizeof(bf16); // 46,080 B
    const size_t head_lds = 16u * HPAD * sizeof(bf16);           //  8,448 B

    dim3 cgrid(NS, 8);
    k_conv<C0P><<<cgrid, 256, conv_lds, stream>>>(in0,  w0b, conv_b0, act0);
    k_conv<HCH><<<cgrid, 256, conv_lds, stream>>>(act0, w1b, conv_b1, act1);

    dim3 hgrid(NS, BATCH / 16);
    k_head<<<hgrid, 256, head_lds, stream>>>(act1, hwb, hb, out);
}
